// _MultiHeadAttention_80814104642164
// MI455X (gfx1250) — matrix operand order repaired and checked by reading, not compiled, not run on hardware
//
#include <hip/hip_runtime.h>

#define B_   4
#define S_   2048
#define DM_  1024
#define H_   16
#define DK_  64
#define SP_  2052   // padded LDS row stride (floats) -> conflict-free tile stores

typedef __attribute__((ext_vector_type(16))) __bf16 v16bf;
typedef __attribute__((ext_vector_type(8)))  __bf16 v8bf;
typedef __attribute__((ext_vector_type(4)))  __bf16 v4bf;
typedef __attribute__((ext_vector_type(8)))  float  v8f;
typedef __attribute__((ext_vector_type(4)))  float  v4f;
typedef __attribute__((ext_vector_type(4)))  unsigned int v4u;
typedef __attribute__((ext_vector_type(4)))  unsigned int u32x4;
typedef __attribute__((ext_vector_type(8)))  int i32x8;
typedef __attribute__((ext_vector_type(4)))  int i32x4;

union ABbf  { v16bf v; v4u u[2]; __bf16 e[16]; };
union Pack8 { v8bf  h; v4u u; };
union Pack4 { v4bf  h; unsigned long long u; };

// ---- WMMA operand builders (CDNA5 16x16x32 bf16 register layouts) ----
__device__ __forceinline__ v16bf ld_bf16_pair(const __bf16* p1, const __bf16* p2) {
  ABbf r;
  r.u[0] = *(const v4u*)p1;
  r.u[1] = *(const v4u*)p2;
  return r.v;
}
__device__ __forceinline__ v16bf ld_bf16_run(const __bf16* p) {
  ABbf r;
  const v4u* q = (const v4u*)p;
  r.u[0] = q[0];
  r.u[1] = q[1];
  return r.v;
}
__device__ __forceinline__ v16bf cvt_f32_pair(const float* p1, const float* p2) {
  ABbf r;
  v4f a0 = *(const v4f*)p1, a1 = *(const v4f*)(p1 + 4);
  v4f b0 = *(const v4f*)p2, b1 = *(const v4f*)(p2 + 4);
#pragma unroll
  for (int i = 0; i < 4; ++i) {
    r.e[i]      = (__bf16)a0[i];
    r.e[4 + i]  = (__bf16)a1[i];
    r.e[8 + i]  = (__bf16)b0[i];
    r.e[12 + i] = (__bf16)b1[i];
  }
  return r.v;
}
__device__ __forceinline__ v8f wmma_bf16(v16bf a, v16bf b, v8f c) {
  return __builtin_amdgcn_wmma_f32_16x16x32_bf16(false, a, false, b, (short)0, c, false, false);
}

// ---- Tensor Data Mover: 2-D tile load Global->LDS (D# per ISA 08_async_tensor §8) ----
// tensor dims == tile dims (no OOB), data_size = 2 bytes, groups 2/3 zero (2-D).
// clang-23 toolchain: 6-arg builtin (g0, g1, g2, g3, zero int32x8, cpol).
__device__ __forceinline__ unsigned lds_off_of(const void* p) {
  return (unsigned)(unsigned long long)(size_t)p;   // flat-LDS aperture keeps offset in addr[31:0]
}
__device__ __forceinline__ void tdm_load_2d(unsigned lds_byte_off, const void* g,
                                            unsigned tile0, unsigned tile1, unsigned stride) {
  unsigned long long ga = (unsigned long long)(size_t)g;
  u32x4 g0 = { 1u,                                   // count=1, user descriptor
               lds_byte_off,                         // lds_addr (bytes)
               (unsigned)(ga & 0xffffffffull),       // global_addr[31:0]
               (unsigned)((ga >> 32) & 0x01ffffffull) | (2u << 30) }; // addr[56:32] | type=2
  i32x8 g1 = { (int)(1u << 16),                      // data_size=1 (2B); no multicast/iterate/pad
               (int)(tile0 << 16),                   // tensor_dim0[15:0] << 16
               (int)(tile1 << 16),                   // tensor_dim0[31:16]=0 | tensor_dim1[15:0]<<16
               (int)(tile0 << 16),                   // tensor_dim1[31:16]=0 | tile_dim0<<16
               (int)tile1,                           // tile_dim1 | tile_dim2=0
               (int)stride,                          // tensor_dim0_stride[31:0] (elements)
               0, 0 };
  i32x4 z4 = {0, 0, 0, 0};
  i32x8 z8 = {0, 0, 0, 0, 0, 0, 0, 0};
  __builtin_amdgcn_tensor_load_to_lds(g0, g1, z4, z4, z8, 0);
}

// ---- Kernel 1: transpose + bf16-convert a 1024x1024 weight: w[k][n] -> wt[n][k] ----
__global__ void k_wtrans(const float* __restrict__ w, __bf16* __restrict__ wt) {
  int idx = blockIdx.x * 256 + threadIdx.x;   // = n*1024 + k (write-coalesced)
  int n = idx >> 10, k = idx & 1023;
  wt[idx] = (__bf16)w[(size_t)k * DM_ + n];
}

// ---- Kernel 1b: elementwise fp32 -> bf16 (activation pre-conversion) ----
__global__ void k_xcvt(const float* __restrict__ x, __bf16* __restrict__ xb) {
  size_t i = ((size_t)blockIdx.x * 256 + threadIdx.x) * 4;
  v4f f = *(const v4f*)(x + i);
  Pack4 p;
#pragma unroll
  for (int j = 0; j < 4; ++j) p.h[j] = (__bf16)f[j];
  *(unsigned long long*)(xb + i) = p.u;
}

// ---- Kernel 2: Q/K projection, swapped orientation: D = W^T (A) x X^T (B) ----
__global__ void k_proj_qk(const __bf16* __restrict__ xb, const __bf16* __restrict__ wt,
                          const float* __restrict__ bias, __bf16* __restrict__ outp) {
  int L = threadIdx.x, hf = L >> 4, lm = L & 15;
  int n0  = blockIdx.x * 32;
  int gs0 = blockIdx.y * 32;
  const __bf16* arow0 = wt + (size_t)(n0 + lm) * DM_;
  const __bf16* arow1 = arow0 + (size_t)16 * DM_;
  const __bf16* brow0 = xb + (size_t)(gs0 + lm) * DM_;
  const __bf16* brow1 = brow0 + (size_t)16 * DM_;
  v8f acc00 = {0.f,0.f,0.f,0.f,0.f,0.f,0.f,0.f}, acc01 = acc00, acc10 = acc00, acc11 = acc00;
#pragma unroll 2
  for (int k0 = 0; k0 < DM_; k0 += 32) {
    v16bf a0 = ld_bf16_pair(arow0 + k0 + hf * 8, arow0 + k0 + 16 + hf * 8);
    v16bf a1 = ld_bf16_pair(arow1 + k0 + hf * 8, arow1 + k0 + 16 + hf * 8);
    v16bf b0 = ld_bf16_pair(brow0 + k0 + hf * 8, brow0 + k0 + 16 + hf * 8);
    v16bf b1 = ld_bf16_pair(brow1 + k0 + hf * 8, brow1 + k0 + 16 + hf * 8);
    acc00 = wmma_bf16(a0, b0, acc00);
    acc01 = wmma_bf16(a0, b1, acc01);
    acc10 = wmma_bf16(a1, b0, acc10);
    acc11 = wmma_bf16(a1, b1, acc11);
  }
  int bb = gs0 / S_;
  v8f accs[2][2] = {{acc00, acc01}, {acc10, acc11}};
#pragma unroll
  for (int i = 0; i < 2; ++i) {
    int nbase = n0 + i * 16 + 8 * hf;
    int h = nbase >> 6, d = nbase & 63;
    const float* bp = bias + nbase;
#pragma unroll
    for (int j = 0; j < 2; ++j) {
      int sl = (gs0 & (S_ - 1)) + j * 16 + lm;
      Pack8 pk;
#pragma unroll
      for (int r = 0; r < 8; ++r) pk.h[r] = (__bf16)(accs[i][j][r] + bp[r]);
      *(v4u*)(outp + ((size_t)(bb * H_ + h) * S_ + sl) * DK_ + d) = pk.u;
    }
  }
}

// ---- Kernel 3: V projection, normal orientation: D = X (A) x W (B) ----
__global__ void k_proj_v(const __bf16* __restrict__ xb, const __bf16* __restrict__ wt,
                         const float* __restrict__ bias, __bf16* __restrict__ vt) {
  int L = threadIdx.x, hf = L >> 4, lm = L & 15;
  int n0  = blockIdx.x * 32;
  int gs0 = blockIdx.y * 32;
  const __bf16* arow0 = xb + (size_t)(gs0 + lm) * DM_;
  const __bf16* arow1 = arow0 + (size_t)16 * DM_;
  const __bf16* brow0 = wt + (size_t)(n0 + lm) * DM_;
  const __bf16* brow1 = brow0 + (size_t)16 * DM_;
  v8f acc00 = {0.f,0.f,0.f,0.f,0.f,0.f,0.f,0.f}, acc01 = acc00, acc10 = acc00, acc11 = acc00;
#pragma unroll 2
  for (int k0 = 0; k0 < DM_; k0 += 32) {
    v16bf a0 = ld_bf16_pair(arow0 + k0 + hf * 8, arow0 + k0 + 16 + hf * 8);
    v16bf a1 = ld_bf16_pair(arow1 + k0 + hf * 8, arow1 + k0 + 16 + hf * 8);
    v16bf b0 = ld_bf16_pair(brow0 + k0 + hf * 8, brow0 + k0 + 16 + hf * 8);
    v16bf b1 = ld_bf16_pair(brow1 + k0 + hf * 8, brow1 + k0 + 16 + hf * 8);
    acc00 = wmma_bf16(a0, b0, acc00);
    acc01 = wmma_bf16(a0, b1, acc01);
    acc10 = wmma_bf16(a1, b0, acc10);
    acc11 = wmma_bf16(a1, b1, acc11);
  }
  int bb = gs0 / S_;
  v8f accs[2][2] = {{acc00, acc01}, {acc10, acc11}};
#pragma unroll
  for (int i = 0; i < 2; ++i) {
    int sbase = (gs0 & (S_ - 1)) + i * 16 + 8 * hf;
#pragma unroll
    for (int j = 0; j < 2; ++j) {
      int n = n0 + j * 16 + lm;
      int h = n >> 6, d = n & 63;
      float bv = bias[n];
      Pack8 pk;
#pragma unroll
      for (int r = 0; r < 8; ++r) pk.h[r] = (__bf16)(accs[i][j][r] + bv);
      *(v4u*)(vt + ((size_t)(bb * H_ + h) * DK_ + d) * S_ + sbase) = pk.u;
    }
  }
}

// ---- Kernel 4: fused scores -> softmax -> attn write -> context ----
// 128 threads (4 waves). Per-wave TDM double buffering for K tiles and V chunks.
// LDS: P 128KB + K stage 16KB + V stage 8KB + scratch  -> 2 WG/WGP.
__global__ void k_attention(const __bf16* __restrict__ q_ws, const __bf16* __restrict__ k_ws,
                            const __bf16* __restrict__ v_t, const unsigned char* __restrict__ mask,
                            float* __restrict__ ctx_out, float* __restrict__ attn_out) {
  extern __shared__ float smem[];
  float* P    = smem;                      // 16 * SP_
  float* red  = smem + 16 * SP_;           // 128
  float* rmax = red + 128;                 // 16
  float* rsum = rmax + 16;                 // 16
  __bf16* kstage = (__bf16*)(rsum + 16);   // 4 waves * 2 * (16*64) = 8192 elems
  __bf16* vstage = kstage + 8192;          // 4 waves * 2 * (16*32) = 4096 elems

  int tid = threadIdx.x;
  int wv  = __builtin_amdgcn_readfirstlane(tid >> 5);   // wave id, forced uniform (SGPR)
  int L   = tid & 31, hf = L >> 4, lm = L & 15;
  int wg  = blockIdx.x;
  int bh  = wg >> 7;
  int qt  = wg & 127;
  int b   = bh >> 4, h = bh & 15;
  int q0  = qt * 16;

  const __bf16* qrow  = q_ws + ((size_t)bh * S_ + q0 + lm) * DK_;
  const __bf16* kbase = k_ws + (size_t)bh * S_ * DK_;

  v16bf qa0 = ld_bf16_pair(qrow + 0  + hf * 8, qrow + 16 + hf * 8);
  v16bf qa1 = ld_bf16_pair(qrow + 32 + hf * 8, qrow + 48 + hf * 8);

  // ---- phase 1: TDM-pipelined scores into LDS (each wave owns k-tiles wv, wv+4, ...) ----
  __bf16* kbuf = kstage + wv * 2048;       // two 1024-elem buffers
  tdm_load_2d(lds_off_of(kbuf), kbase + (size_t)(wv * 16) * DK_, 64, 16, 64);
  int par = 0;
  int kt;
  for (kt = wv; kt + 4 < S_ / 16; kt += 4) {
    tdm_load_2d(lds_off_of(kbuf + (par ^ 1) * 1024),
                kbase + (size_t)((kt + 4) * 16) * DK_, 64, 16, 64);
    __builtin_amdgcn_s_wait_tensorcnt(1);
    const __bf16* kr = kbuf + par * 1024 + lm * 64;
    v8f acc = {0.f,0.f,0.f,0.f,0.f,0.f,0.f,0.f};
    acc = wmma_bf16(qa0, ld_bf16_pair(kr + 0  + hf * 8, kr + 16 + hf * 8), acc);
    acc = wmma_bf16(qa1, ld_bf16_pair(kr + 32 + hf * 8, kr + 48 + hf * 8), acc);
    int col = kt * 16 + lm;
#pragma unroll
    for (int r = 0; r < 8; ++r) P[(size_t)(r + 8 * hf) * SP_ + col] = acc[r] * 0.125f;
    par ^= 1;
  }
  __builtin_amdgcn_s_wait_tensorcnt(0);
  {
    const __bf16* kr = kbuf + par * 1024 + lm * 64;
    v8f acc = {0.f,0.f,0.f,0.f,0.f,0.f,0.f,0.f};
    acc = wmma_bf16(qa0, ld_bf16_pair(kr + 0  + hf * 8, kr + 16 + hf * 8), acc);
    acc = wmma_bf16(qa1, ld_bf16_pair(kr + 32 + hf * 8, kr + 48 + hf * 8), acc);
    int col = kt * 16 + lm;
#pragma unroll
    for (int r = 0; r < 8; ++r) P[(size_t)(r + 8 * hf) * SP_ + col] = acc[r] * 0.125f;
  }
  // prefetch first V chunk under the softmax phase
  int d0c = wv * 16;
  const __bf16* vsrc = v_t + ((size_t)bh * DK_ + d0c) * S_;
  __bf16* vbuf = vstage + wv * 1024;       // two 512-elem buffers
  tdm_load_2d(lds_off_of(vbuf), vsrc, 32, 16, S_);
  __syncthreads();

  // ---- phase 2: mask + numerically stable softmax (8 threads/row, 256 cols each) ----
  int row = tid >> 3;
  int cs  = (tid & 7) * 256;
  const unsigned char* mrow = mask + ((size_t)b * S_ + q0 + row) * S_ + cs;
  float lmax = -3.0e38f;
  for (int i = 0; i < 256; ++i) {
    float p = P[(size_t)row * SP_ + cs + i];
    if (mrow[i]) p = -1.0e9f;
    P[(size_t)row * SP_ + cs + i] = p;
    lmax = fmaxf(lmax, p);
  }
  red[tid] = lmax;
  __syncthreads();
  if (tid < 16) {
    float m = red[tid * 8];
#pragma unroll
    for (int j = 1; j < 8; ++j) m = fmaxf(m, red[tid * 8 + j]);
    rmax[tid] = m;
  }
  __syncthreads();
  float rm = rmax[row];
  float lsum = 0.f;
  for (int i = 0; i < 256; ++i) {
    float e = __expf(P[(size_t)row * SP_ + cs + i] - rm);
    P[(size_t)row * SP_ + cs + i] = e;
    lsum += e;
  }
  red[tid] = lsum;
  __syncthreads();
  if (tid < 16) {
    float s = 0.f;
#pragma unroll
    for (int j = 0; j < 8; ++j) s += red[tid * 8 + j];
    rsum[tid] = 1.0f / s;
  }
  __syncthreads();

  // ---- normalize in LDS + coalesced NON-TEMPORAL attn write (1.07 GB stream) ----
  float* arow_out = attn_out + ((size_t)bh * S_ + q0) * S_;
  for (int r2 = 0; r2 < 16; ++r2) {
    float inv = rsum[r2];
#pragma unroll
    for (int j = 0; j < 4; ++j) {
      int c = (tid + j * 128) * 4;
      v4f v = *(v4f*)&P[(size_t)r2 * SP_ + c];
      v = v * inv;
      *(v4f*)&P[(size_t)r2 * SP_ + c] = v;
      __builtin_nontemporal_store(v, (v4f*)(arow_out + (size_t)r2 * S_ + c));
    }
  }
  __syncthreads();

  // ---- phase 3: TDM-pipelined context = P(16xS) x V(Sx64); wave wv owns d'-tile wv ----
  const float* Prow = P + (size_t)lm * SP_;
  v8f acc = {0.f,0.f,0.f,0.f,0.f,0.f,0.f,0.f};
  int p3 = 0;
  int c;
  for (c = 0; c + 1 < S_ / 32; ++c) {
    tdm_load_2d(lds_off_of(vbuf + (p3 ^ 1) * 512), vsrc + (c + 1) * 32, 32, 16, S_);
    __builtin_amdgcn_s_wait_tensorcnt(1);
    int c0 = c * 32;
    v16bf a   = cvt_f32_pair(Prow + c0 + hf * 8, Prow + c0 + 16 + hf * 8);
    v16bf bb2 = ld_bf16_pair(vbuf + p3 * 512 + lm * 32 + hf * 8, vbuf + p3 * 512 + lm * 32 + 16 + hf * 8);
    acc = wmma_bf16(a, bb2, acc);
    p3 ^= 1;
  }
  __builtin_amdgcn_s_wait_tensorcnt(0);
  {
    int c0 = c * 32;
    v16bf a   = cvt_f32_pair(Prow + c0 + hf * 8, Prow + c0 + 16 + hf * 8);
    v16bf bb2 = ld_bf16_pair(vbuf + p3 * 512 + lm * 32 + hf * 8, vbuf + p3 * 512 + lm * 32 + 16 + hf * 8);
    acc = wmma_bf16(a, bb2, acc);
  }
  float* ctx = ctx_out + ((size_t)b * S_ + q0) * (H_ * DK_) + h * DK_;
#pragma unroll
  for (int r = 0; r < 8; ++r)
    ctx[(size_t)(r + 8 * hf) * (H_ * DK_) + d0c + lm] = acc[r];
}

extern "C" void kernel_launch(void* const* d_in, const int* in_sizes, int n_in,
                              void* d_out, int out_size, void* d_ws, size_t ws_size,
                              hipStream_t stream) {
  const float* q  = (const float*)d_in[0];
  const float* k  = (const float*)d_in[1];
  const float* v  = (const float*)d_in[2];
  const unsigned char* mask = (const unsigned char*)d_in[3];
  const float* w_q = (const float*)d_in[4];
  const float* w_k = (const float*)d_in[5];
  const float* w_v = (const float*)d_in[6];
  const float* b_q = (const float*)d_in[7];
  const float* b_k = (const float*)d_in[8];
  const float* b_v = (const float*)d_in[9];

  float* ctx  = (float*)d_out;                               // B*S*1024
  float* attn = (float*)d_out + (size_t)B_ * S_ * DM_;       // B*H*S*S

  const size_t NW = (size_t)DM_ * DM_;      // 1M  (weight elems)
  const size_t NX = (size_t)B_ * S_ * DM_;  // 8M  (activation elems)

  __bf16* wt_q = (__bf16*)d_ws;
  __bf16* wt_k = wt_q + NW;
  __bf16* wt_v = wt_k + NW;
  __bf16* xq   = wt_v + NW;
  __bf16* xk   = xq + NX;
  __bf16* xv   = xk + NX;
  __bf16* q_ws = xv + NX;
  __bf16* k_ws = q_ws + NX;
  __bf16* v_tw = k_ws + NX;

  k_wtrans<<<(int)(NW / 256), 256, 0, stream>>>(w_q, wt_q);
  k_wtrans<<<(int)(NW / 256), 256, 0, stream>>>(w_k, wt_k);
  k_wtrans<<<(int)(NW / 256), 256, 0, stream>>>(w_v, wt_v);

  k_xcvt<<<(int)(NX / 1024), 256, 0, stream>>>(q, xq);
  k_xcvt<<<(int)(NX / 1024), 256, 0, stream>>>(k, xk);
  k_xcvt<<<(int)(NX / 1024), 256, 0, stream>>>(v, xv);

  dim3 pgrid(DM_ / 32, B_ * S_ / 32);   // 2x2-blocked tiles
  k_proj_qk<<<pgrid, 32, 0, stream>>>(xq, wt_q, b_q, q_ws);
  k_proj_qk<<<pgrid, 32, 0, stream>>>(xk, wt_k, b_k, k_ws);
  k_proj_v <<<pgrid, 32, 0, stream>>>(xv, wt_v, b_v, v_tw);

  int smem_bytes = (16 * SP_ + 128 + 16 + 16) * (int)sizeof(float)
                 + (8192 + 4096) * (int)sizeof(__bf16);      // ~153 KB
  k_attention<<<B_ * H_ * (S_ / 16), 128, smem_bytes, stream>>>(q_ws, k_ws, v_tw, mask, ctx, attn);
}
